// Decoder_23828478558525
// MI455X (gfx1250) — matrix operand order repaired and checked by reading, not compiled, not run on hardware
//
#include <hip/hip_runtime.h>
#include <hip/hip_bf16.h>
#include <math.h>

// Problem constants (reference): B=512, N=200, D=256, H=8, DK=32
#define NEG_ (-1e9f)

typedef __attribute__((ext_vector_type(16))) __bf16 v16bf;
typedef __attribute__((ext_vector_type(8)))  float  v8f;

__device__ __forceinline__ unsigned short f2bf(float f) {
  union { float f; unsigned u; } x; x.f = f;
  unsigned r = x.u + 0x7FFFu + ((x.u >> 16) & 1u);  // RNE
  return (unsigned short)(r >> 16);
}
__device__ __forceinline__ float bf2f(unsigned short h) {
  union { unsigned u; float f; } x; x.u = ((unsigned)h) << 16;
  return x.f;
}

// ---------------------------------------------------------------------------
// One-time f32 -> bf16 conversion of embeddings (A panel for the WMMA GEMM).
// ---------------------------------------------------------------------------
__global__ void conv_emb(const float* __restrict__ A, unsigned short* __restrict__ Ab,
                         int nchunks) {
  int idx = blockIdx.x * blockDim.x + threadIdx.x;  // chunk of 8 elements
  if (idx >= nchunks) return;
  const float4* p = (const float4*)(A + (size_t)idx * 8);
  float4 a = p[0], b = p[1];
  union { unsigned short u[8]; uint4 v; } o;
  o.u[0] = f2bf(a.x); o.u[1] = f2bf(a.y); o.u[2] = f2bf(a.z); o.u[3] = f2bf(a.w);
  o.u[4] = f2bf(b.x); o.u[5] = f2bf(b.y); o.u[6] = f2bf(b.z); o.u[7] = f2bf(b.w);
  *(uint4*)(Ab + (size_t)idx * 8) = o.v;
}

// ---------------------------------------------------------------------------
// Prep: build transposed, folded weight panel BcatT[1024][256] (bf16).
// cols 0..255   : W_v  = W_node[:,256:512]          (glimpse K==V per ref)
// cols 256..511 : G^T  = W_l @ W_out^T  (W_out folded into logit keys)
// cols 512..767 : W_step[:256,:]   (E1: first-node context)
// cols 768..1023: W_step[256:,:]   (E2: current-node context)
// ---------------------------------------------------------------------------
__global__ void prep_bcat(const float* __restrict__ Wn, const float* __restrict__ Ws,
                          const float* __restrict__ Wo, unsigned short* __restrict__ bT) {
  int c = blockIdx.x, k = threadIdx.x;  // c: output col, k: reduction row
  float v;
  if (c < 256) {
    v = Wn[k * 768 + 256 + c];
  } else if (c < 512) {
    int i = c - 256;
    float acc = 0.f;
    for (int j = 0; j < 256; ++j) acc += Wn[k * 768 + 512 + j] * Wo[i * 256 + j];
    v = acc;
  } else if (c < 768) {
    v = Ws[k * 256 + (c - 512)];
  } else {
    v = Ws[(256 + k) * 256 + (c - 768)];
  }
  bT[(size_t)c * 256 + k] = f2bf(v);
}

// q0 = W_placeholder @ W_step  (t==0 step context; constant across batch)
__global__ void prep_q0(const float* __restrict__ Wp, const float* __restrict__ Ws,
                        float* __restrict__ q0) {
  int j = threadIdx.x;
  float acc = 0.f;
  for (int i = 0; i < 512; ++i) acc += Wp[i] * Ws[i * 256 + j];
  q0[j] = acc;
}

// graph_context[b] = mean_n(emb[b]) @ W_fixed
__global__ void prep_gc(const float* __restrict__ emb, const float* __restrict__ Wf,
                        float* __restrict__ gc) {
  __shared__ float mean[256];
  int b = blockIdx.x, k = threadIdx.x;
  const float* e = emb + (size_t)b * 200 * 256 + k;
  float s = 0.f;
  for (int n = 0; n < 200; ++n) s += e[(size_t)n * 256];
  mean[k] = s * (1.0f / 200.0f);
  __syncthreads();
  float acc = 0.f;
  for (int i = 0; i < 256; ++i) acc += mean[i] * Wf[i * 256 + k];
  gc[b * 256 + k] = acc;
}

// ---------------------------------------------------------------------------
// LDS-staged WMMA GEMM: C[102400,1024](bf16) = Abf[102400,256](bf16) @ Bcat.
// Block (256 thr / 8 waves) owns one 16-row M tile x 256 N cols.
// A slab (16x256, 8KB) and B slab (256x256, 132KB padded) staged in LDS once;
// K-loop is ds_load_b128 + v_wmma only (no VMEM latency in the hot loop).
// Wave w computes 2 N tiles -> 16 v_wmma per wave.
// Fragment layouts (wave32): A/B use contiguous K per lane half (K = 16*g + s); C/D per CDNA5 ISA.
// ---------------------------------------------------------------------------
#define GP_STRIDE 264  // row pad: 132 dwords -> conflict-free 16-lane access

__global__ void gemm_pre(const unsigned short* __restrict__ Ab,
                         const unsigned short* __restrict__ bT,
                         unsigned short* __restrict__ C) {
  extern __shared__ unsigned char gsm[];
  unsigned short* Blds = (unsigned short*)gsm;          // [256][264] bf16
  unsigned short* Alds = Blds + 256 * GP_STRIDE;        // [16][264] bf16

  int tid = threadIdx.x;
  int wave = tid >> 5, lane = tid & 31;
  int i = blockIdx.y;                 // M tile (0..6399)
  int jtBase = blockIdx.x * 16;       // first of 16 N tiles owned by block
  int colBase = jtBase * 16;          // first of 256 bT rows (cols of Bcat)
  int m = lane & 15, g = lane >> 4;   // row-in-tile, lane-half

  // Stage B slab: 256 rows x 256 bf16 (32 x 16B chunks per row)
  for (int idx = tid; idx < 256 * 32; idx += 256) {
    int c = idx >> 5, q = (idx & 31) * 8;
    *(uint4*)&Blds[c * GP_STRIDE + q] =
        *(const uint4*)(bT + (size_t)(colBase + c) * 256 + q);
  }
  // Stage A slab: 16 rows x 256 bf16
  for (int idx = tid; idx < 16 * 32; idx += 256) {
    int r = idx >> 5, q = (idx & 31) * 8;
    *(uint4*)&Alds[r * GP_STRIDE + q] =
        *(const uint4*)(Ab + (size_t)(i * 16 + r) * 256 + q);
  }
  __syncthreads();

  int t0 = wave * 2, t1 = wave * 2 + 1;   // this wave's N tiles (rel.)
  const unsigned short* ArL  = Alds + m * GP_STRIDE;
  const unsigned short* Br0L = Blds + (t0 * 16 + m) * GP_STRIDE;  // n == lane&15
  const unsigned short* Br1L = Blds + (t1 * 16 + m) * GP_STRIDE;
  v8f acc0 = {0.f,0.f,0.f,0.f,0.f,0.f,0.f,0.f};
  v8f acc1 = acc0;
#pragma unroll
  for (int kk = 0; kk < 256; kk += 32) {
    union { unsigned short u[16]; v16bf v; } Af, B0, B1;
    // A 16x32 frag: lane half g holds K = g*16+[0..15]
    *(uint4*)&Af.u[0] = *(const uint4*)(ArL + kk + g * 16);
    *(uint4*)&Af.u[8] = *(const uint4*)(ArL + kk + g * 16 + 8);
    // B 32x16 frags: lane half g holds K = g*16+[0..15] (pre-transposed)
    int bo = kk + g * 16;
    *(uint4*)&B0.u[0] = *(const uint4*)(Br0L + bo);
    *(uint4*)&B0.u[8] = *(const uint4*)(Br0L + bo + 8);
    *(uint4*)&B1.u[0] = *(const uint4*)(Br1L + bo);
    *(uint4*)&B1.u[8] = *(const uint4*)(Br1L + bo + 8);
    acc0 = __builtin_amdgcn_wmma_f32_16x16x32_bf16(false, Af.v, false, B0.v, (short)0, acc0, false, false);
    acc1 = __builtin_amdgcn_wmma_f32_16x16x32_bf16(false, Af.v, false, B1.v, (short)0, acc1, false, false);
  }
  // C/D 16x16 f32: VGPR r -> M = r + 8*g, N = lane&15
  unsigned short* Crow = C + (size_t)(i * 16) * 1024 + m;
#pragma unroll
  for (int r = 0; r < 8; ++r) {
    size_t ro = (size_t)(r + g * 8) * 1024;
    Crow[ro + (jtBase + t0) * 16] = f2bf(acc0[r]);
    Crow[ro + (jtBase + t1) * 16] = f2bf(acc1[r]);
  }
}

// ---------------------------------------------------------------------------
// Persistent decode: 1 workgroup (256 thr / 8 waves) per batch element.
// KV + folded logit keys live in LDS (bf16) for all 200 steps.
// ---------------------------------------------------------------------------
#define SM_STRIDE 264  // 256 + 8 bf16 pad (row stride 528B, 16B aligned)

__global__ void decode(const unsigned short* __restrict__ Cp,
                       const float* __restrict__ gc,
                       const float* __restrict__ q0,
                       float* __restrict__ outLP, float* __restrict__ outAct) {
  extern __shared__ unsigned char smem[];
  unsigned short* kv = (unsigned short*)smem;           // [200][264] bf16
  unsigned short* lk = kv + 200 * SM_STRIDE;            // [200][264] bf16
  float* qs  = (float*)(lk + 200 * SM_STRIDE);          // [256] scaled query
  float* gs  = qs + 256;                                // [256] scaled glimpse
  float* gcl = gs + 256;                                // [256] graph context
  float* sc  = gcl + 256;                               // [8][200] scores/attn
  float* lg  = sc + 8 * 200;                            // [200] logits
  int*   vis = (int*)(lg + 200);                        // [200] visited
  int*   ctl = vis + 200;                               // [0]=first [1]=cur
  float* ctf = (float*)(ctl + 4);                       // [0]=max [1]=log(sum)

  int b = blockIdx.x, tid = threadIdx.x;
  int lane = tid & 31, wv = tid >> 5;
  size_t rowBase = (size_t)b * 200 * 1024;

  // init: visited, control, graph ctx, load KV/LK tiles into LDS (16B chunks)
  for (int n = tid; n < 200; n += 256) vis[n] = 0;
  if (tid == 0) { ctl[0] = 0; ctl[1] = 0; }
  gcl[tid] = gc[b * 256 + tid];
  for (int idx = tid; idx < 6400; idx += 256) {
    int n = idx >> 5, c8 = (idx & 31) * 8;
    const unsigned short* src = Cp + rowBase + (size_t)n * 1024 + c8;
    *(uint4*)&kv[n * SM_STRIDE + c8] = *(const uint4*)(src);
    *(uint4*)&lk[n * SM_STRIDE + c8] = *(const uint4*)(src + 256);
  }
  __syncthreads();

  const float sscale = 0.17677669529663689f;  // 1/sqrt(DK=32)
  const float lscale = 0.0625f;               // 1/sqrt(D=256)

  for (int t = 0; t < 200; ++t) {
    // A: q = graph_ctx + step_ctx (gathers from precomputed E1/E2 panels)
    {
      float v = gcl[tid];
      if (t == 0) {
        v += q0[tid];
      } else {
        int f = ctl[0], cu = ctl[1];
        v += bf2f(Cp[rowBase + (size_t)f  * 1024 + 512 + tid]);
        v += bf2f(Cp[rowBase + (size_t)cu * 1024 + 768 + tid]);
      }
      qs[tid] = v * sscale;
    }
    __syncthreads();
    // B: scores[h][n] = (q_h . KV[n,h]) / sqrt(DK), masked
    for (int idx = tid; idx < 1600; idx += 256) {
      int h = idx & 7, n = idx >> 3;
      const unsigned short* kr = &kv[n * SM_STRIDE + h * 32];
      const float* qr = &qs[h * 32];
      float acc = 0.f;
#pragma unroll
      for (int k2 = 0; k2 < 32; ++k2) acc += qr[k2] * bf2f(kr[k2]);
      sc[h * 200 + n] = vis[n] ? NEG_ : acc;
    }
    __syncthreads();
    // C: softmax per head (wave wv handles head wv; wave32 shuffles)
    {
      float m = -3.4e38f;
      for (int n = lane; n < 200; n += 32) m = fmaxf(m, sc[wv * 200 + n]);
#pragma unroll
      for (int off = 16; off >= 1; off >>= 1) m = fmaxf(m, __shfl_xor(m, off));
      float s = 0.f;
      for (int n = lane; n < 200; n += 32) s += __expf(sc[wv * 200 + n] - m);
#pragma unroll
      for (int off = 16; off >= 1; off >>= 1) s += __shfl_xor(s, off);
      float rinv = 1.0f / s;
      for (int n = lane; n < 200; n += 32)
        sc[wv * 200 + n] = __expf(sc[wv * 200 + n] - m) * rinv;
    }
    __syncthreads();
    // D: glimpse (pre-projection; W_out already folded into lk)
    {
      int h = tid >> 5;
      const float* an = &sc[h * 200];
      float acc = 0.f;
#pragma unroll 4
      for (int n = 0; n < 200; ++n) acc += an[n] * bf2f(kv[n * SM_STRIDE + tid]);
      gs[tid] = acc * lscale;
    }
    __syncthreads();
    // E: logits[n] = 10*tanh( g . LK2[n] / sqrt(D) ), masked
    if (tid < 200) {
      const unsigned short* lr = &lk[tid * SM_STRIDE];
      float acc = 0.f;
#pragma unroll 8
      for (int j = 0; j < 256; ++j) acc += gs[j] * bf2f(lr[j]);
      float l = 10.0f * tanhf(acc);
      lg[tid] = vis[tid] ? NEG_ : l;
    }
    __syncthreads();
    // F: wave 0: argmax (first-occurrence ties) + log-softmax stats + state
    if (wv == 0) {
      float bv = -3.4e38f; int bi = 0;
      for (int n = lane; n < 200; n += 32) {
        float v = lg[n];
        if (v > bv) { bv = v; bi = n; }
      }
#pragma unroll
      for (int off = 16; off >= 1; off >>= 1) {
        float ov = __shfl_xor(bv, off); int oi = __shfl_xor(bi, off);
        if (ov > bv || (ov == bv && oi < bi)) { bv = ov; bi = oi; }
      }
      float s = 0.f;
      for (int n = lane; n < 200; n += 32) s += __expf(lg[n] - bv);
#pragma unroll
      for (int off = 16; off >= 1; off >>= 1) s += __shfl_xor(s, off);
      if (lane == 0) {
        ctf[0] = bv; ctf[1] = __logf(s);
        vis[bi] = 1; ctl[1] = bi;
        if (t == 0) ctl[0] = bi;
        outAct[b * 200 + t] = (float)bi;  // actions emitted as float
      }
    }
    __syncthreads();
    // G: write log_p row
    {
      float mval = ctf[0], lsum = ctf[1];
      float* orow = outLP + (size_t)b * 40000 + (size_t)t * 200;
      for (int n = tid; n < 200; n += 256) orow[n] = lg[n] - mval - lsum;
    }
    __syncthreads();
  }
}

extern "C" void kernel_launch(void* const* d_in, const int* in_sizes, int n_in,
                              void* d_out, int out_size, void* d_ws, size_t ws_size,
                              hipStream_t stream) {
  const float* emb = (const float*)d_in[0];  // [512,200,256]
  const float* Wn  = (const float*)d_in[1];  // [256,768]
  const float* Wf  = (const float*)d_in[2];  // [256,256]
  const float* Ws  = (const float*)d_in[3];  // [512,256]
  const float* Wp  = (const float*)d_in[4];  // [512]
  const float* Wo  = (const float*)d_in[5];  // [256,256]
  float* out = (float*)d_out;                // [512*200*200] log_p + [512*200] actions

  unsigned char* wsb = (unsigned char*)d_ws;
  unsigned short* bT = (unsigned short*)wsb;        wsb += (size_t)1024 * 256 * 2;
  float* q0 = (float*)wsb;                          wsb += (size_t)256 * 4;
  float* gc = (float*)wsb;                          wsb += (size_t)512 * 256 * 4;
  unsigned short* Abf = (unsigned short*)wsb;       wsb += (size_t)102400 * 256 * 2;
  size_t off = (size_t)(wsb - (unsigned char*)d_ws);
  off = (off + 255) & ~(size_t)255;
  unsigned short* Cp = (unsigned short*)((unsigned char*)d_ws + off);  // [102400][1024] bf16

  const int nchunks = 102400 * 256 / 8;  // 3,276,800 chunks of 8 elems
  conv_emb <<<dim3((nchunks + 255) / 256), dim3(256), 0, stream>>>(emb, Abf, nchunks);
  prep_bcat<<<dim3(1024), dim3(256), 0, stream>>>(Wn, Ws, Wo, bT);
  prep_q0  <<<dim3(1),    dim3(256), 0, stream>>>(Wp, Ws, q0);
  prep_gc  <<<dim3(512),  dim3(256), 0, stream>>>(emb, Wf, gc);

  size_t smemG = (size_t)(256 + 16) * GP_STRIDE * 2;  // B slab + A slab (bf16)
  hipFuncSetAttribute((const void*)gemm_pre,
                      hipFuncAttributeMaxDynamicSharedMemorySize, (int)smemG);
  gemm_pre <<<dim3(4, 6400), dim3(256), smemG, stream>>>(Abf, bT, Cp);

  size_t smem = (size_t)2 * 200 * SM_STRIDE * 2           // kv + lk (bf16)
              + (size_t)(256 * 3 + 8 * 200 + 200) * 4     // qs/gs/gcl/sc/lg
              + (size_t)200 * 4 + 32;                     // vis + ctl/ctf
  hipFuncSetAttribute((const void*)decode,
                      hipFuncAttributeMaxDynamicSharedMemorySize, (int)smem);
  decode<<<dim3(512), dim3(256), smem, stream>>>(Cp, gc, q0,
                                                 out, out + (size_t)512 * 200 * 200);
}
